// RNN_17678085390657
// MI455X (gfx1250) — matrix operand order repaired and checked by reading, not compiled, not run on hardware
//
#include <hip/hip_runtime.h>
#include <math.h>

// Problem constants (from reference): B=64, T=512, I=512, H=1024
#define B_    64
#define T_    512
#define I_    512
#define H_    1024
#define KTOT  1536            // H_ + I_  (fused [h | x] K dimension)
#define FOURH 4096
#define LDS_PAD_BF16 8
#define LDS_STRIDE ((KTOT + LDS_PAD_BF16) * 2)   // bytes per A row in LDS = 3088

typedef __attribute__((ext_vector_type(16))) __bf16 v16bf;
typedef __attribute__((ext_vector_type(8)))  float  v8f;

union Frag {
  v16bf v;
  uint4 u[2];
};

__device__ __forceinline__ unsigned short f2bf(float f) {
  unsigned u = __builtin_bit_cast(unsigned, f);
  u += 0x7FFFu + ((u >> 16) & 1u);   // round-to-nearest-even
  return (unsigned short)(u >> 16);
}

// ---- fast transcendental gates: native v_tanh_f32 when available ----
__device__ __forceinline__ float fast_tanh(float z) {
#if __has_builtin(__builtin_amdgcn_tanhf)
  return __builtin_amdgcn_tanhf(z);
#elif __has_builtin(__builtin_amdgcn_tanh_f32)
  return __builtin_amdgcn_tanh_f32(z);
#else
  // tanh(z) = 1 - 2/(1 + exp(2z)); v_exp_f32 + v_rcp_f32, branch-free
  float e = __builtin_amdgcn_exp2f(z * 2.8853900817779268f);  // 2*log2(e)
  return 1.0f - 2.0f * __builtin_amdgcn_rcpf(1.0f + e);
#endif
}

// ---- async global->LDS (CDNA5) with plain-copy fallback ----
#if __has_builtin(__builtin_amdgcn_global_load_async_to_lds_b128)
#define HAVE_ASYNC_LDS 1
typedef int v4i_vs __attribute__((vector_size(16)));
typedef __attribute__((address_space(1))) v4i_vs* g_v4i_ptr;   // pointee in AS1 (global)
typedef __attribute__((address_space(3))) v4i_vs* l_v4i_ptr;   // pointee in AS3 (LDS)
__device__ __forceinline__ void async_cp16(const void* g, void* l) {
  __builtin_amdgcn_global_load_async_to_lds_b128((g_v4i_ptr)(void*)g,
                                                 (l_v4i_ptr)l, 0, 0);
}
__device__ __forceinline__ void wait_async0() {
#if __has_builtin(__builtin_amdgcn_s_wait_asynccnt)
  __builtin_amdgcn_s_wait_asynccnt(0);
#else
  asm volatile("s_wait_asynccnt 0" ::: "memory");
#endif
}
#endif

// -------------------------------------------------------------------------
// One-time: build fused, transposed bf16 weights.
// Wct[n][k], n in [0,4096) (z column), k in [0,1536):
//   k <  1024 : Wh[k][n] ;  k >= 1024 : Wx[k-1024][n]
// -------------------------------------------------------------------------
__global__ __launch_bounds__(256) void cvt_weights(const float* __restrict__ Wx,
                                                   const float* __restrict__ Wh,
                                                   unsigned short* __restrict__ Wct) {
  unsigned gid = blockIdx.x * 256u + threadIdx.x;   // 4096*1536 total
  if (gid >= (unsigned)FOURH * KTOT) return;
  unsigned k = gid / FOURH;          // consecutive threads -> consecutive n (coalesced read)
  unsigned n = gid % FOURH;
  float v = (k < H_) ? Wh[(size_t)k * FOURH + n]
                     : Wx[(size_t)(k - H_) * FOURH + n];
  Wct[(size_t)n * KTOT + k] = f2bf(v);
}

// One-time: whole x tensor fp32 -> bf16 (read once per step thereafter).
__global__ __launch_bounds__(256) void cvt_x(const float* __restrict__ x,
                                             unsigned short* __restrict__ xb) {
  size_t gid = (size_t)blockIdx.x * 256 + threadIdx.x;   // one float4 each
  if (gid >= (size_t)B_ * T_ * I_ / 4) return;
  float4 v = ((const float4*)x)[gid];
  uint2 st;
  st.x = (unsigned)f2bf(v.x) | ((unsigned)f2bf(v.y) << 16);
  st.y = (unsigned)f2bf(v.z) | ((unsigned)f2bf(v.w) << 16);
  ((uint2*)xb)[gid] = st;
}

// One-time: h0 -> bf16 ping buffer, c0 -> fp32 cell buffer.
__global__ __launch_bounds__(256) void init_state(const float* __restrict__ h0,
                                                  const float* __restrict__ c0,
                                                  unsigned short* __restrict__ hbuf0,
                                                  float* __restrict__ cbuf) {
  unsigned i = blockIdx.x * 256u + threadIdx.x;
  if (i < B_ * H_) {
    hbuf0[i] = f2bf(h0[i]);
    cbuf[i]  = c0[i];
  }
}

// -------------------------------------------------------------------------
// One LSTM timestep.
// grid = (64 hidden tiles j, 4 row groups r), block = 128 (4 waves).
// Wave g computes gate g's 16x16 z-tile via 48x v_wmma_f32_16x16x32_bf16,
// then all waves fuse the gate nonlinearities + c/h update through LDS.
// -------------------------------------------------------------------------
__global__ __launch_bounds__(128) void lstm_step(const unsigned short* __restrict__ xb,
                                                 const unsigned short* __restrict__ Wct,
                                                 const float* __restrict__ bias,
                                                 const unsigned short* __restrict__ hread,
                                                 unsigned short* __restrict__ hwrite,
                                                 float* __restrict__ cbuf,
                                                 float* __restrict__ out,
                                                 int t, int is_last) {
  __shared__ __align__(16) unsigned char ldsA[16 * LDS_STRIDE];  // A = [h|x] slab, bf16
  __shared__ float ldsG[4 * 256];                                // 4 activated gate tiles

  const int tid  = threadIdx.x;
  const int lane = tid & 31;
  const int wave = tid >> 5;           // gate index 0..3 (i,f,g,o)
  const int j    = blockIdx.x;         // hidden tile 0..63
  const int r    = blockIdx.y;         // row group 0..3

  // ---- Stage A rows [r*16, r*16+16) into LDS (pure 16B copies) ----
  // h part: 16 rows x 1024 bf16, contiguous in global: 2048 uint4.
  {
    const uint4* src = (const uint4*)(hread + (size_t)r * 16 * H_);
    for (int idx = tid; idx < 16 * 128; idx += 128) {
      int row = idx >> 7, ck = idx & 127;
      void* dst = ldsA + row * LDS_STRIDE + ck * 16;
#ifdef HAVE_ASYNC_LDS
      async_cp16(src + idx, dst);
#else
      *(uint4*)dst = src[idx];
#endif
    }
  }
  // x part: row b = r*16+m at xb[(b*T + t)*I], 64 uint4 per row.
  {
    for (int idx = tid; idx < 16 * 64; idx += 128) {
      int row = idx >> 6, ck = idx & 63;
      const uint4* sp = (const uint4*)(xb + ((size_t)(r * 16 + row) * T_ + t) * I_) + ck;
      void* dst = ldsA + row * LDS_STRIDE + H_ * 2 + ck * 16;
#ifdef HAVE_ASYNC_LDS
      async_cp16(sp, dst);
#else
      *(uint4*)dst = *sp;
#endif
    }
  }
#ifdef HAVE_ASYNC_LDS
  wait_async0();
#endif
  __syncthreads();

  // ---- GEMM: 16x16 tile, K = 1536 ----
  const int n0 = wave * H_ + j * 16;                      // z column base for this gate
  const int laneHi = lane >> 4;                           // 0: lanes 0-15, 1: lanes 16-31
  // A fragment (ISA 16-bit A 16x32): lane<16 -> K {k0..+7, k0+16..+23}
  const unsigned char* aRow = ldsA + (lane & 15) * LDS_STRIDE + (laneHi << 4);
  // B fragment (ISA 32x16 B): lane n holds column n, K {k0..+7, k0+16..+23} (+8 for lanes 16-31)
  const unsigned char* bRow = (const unsigned char*)(Wct + (size_t)(n0 + (lane & 15)) * KTOT)
                              + (laneHi << 4);

  v8f acc = {};
#pragma unroll 4
  for (int k0 = 0; k0 < KTOT; k0 += 32) {
    Frag fa, fb;
    const unsigned char* ab = aRow + k0 * 2;
    fa.u[0] = *(const uint4*)(ab);
    fa.u[1] = *(const uint4*)(ab + 32);
    const unsigned char* bb = bRow + k0 * 2;
    fb.u[0] = *(const uint4*)(bb);
    fb.u[1] = *(const uint4*)(bb + 32);
    acc = __builtin_amdgcn_wmma_f32_16x16x32_bf16(false, fa.v, false, fb.v,
                                                  (short)0, acc, false, false);
  }

  // ---- bias + per-gate activation (branch-free, 1 trans op/elem) ----
  // gate g (wave 2): tanh(z).  gates i,f,o: sigmoid(z) = 0.5*tanh(0.5z)+0.5
  float bv = bias[n0 + (lane & 15)];
  float sc = (wave == 2) ? 1.0f : 0.5f;   // pre-scale
  float ad = (wave == 2) ? 0.0f : 0.5f;   // post-offset (and post-scale == sc)
#pragma unroll
  for (int v = 0; v < 8; ++v) {
    float z = acc[v] + bv;
    float a = __builtin_fmaf(sc, fast_tanh(z * sc), ad);
    ldsG[wave * 256 + v * 32 + lane] = a;
  }
  __syncthreads();

  // ---- fused c/h update: 2 elements per thread of the 16x16 tile ----
#pragma unroll
  for (int e2 = 0; e2 < 2; ++e2) {
    int e = tid * 2 + e2;
    int m = e >> 4, n = e & 15;
    int li = (m & 7) * 32 + ((m >> 3) << 4) + n;          // C-layout (VGPR,lane) index
    float gi = ldsG[0 * 256 + li];
    float gf = ldsG[1 * 256 + li];
    float gg = ldsG[2 * 256 + li];
    float go = ldsG[3 * 256 + li];
    size_t cidx = (size_t)(r * 16 + m) * H_ + j * 16 + n;
    float cn = __builtin_fmaf(gf, cbuf[cidx], gi * gg);
    cbuf[cidx] = cn;
    float hn = go * fast_tanh(cn);
    hwrite[cidx] = f2bf(hn);
    if (is_last) out[cidx] = hn;
  }
}

// -------------------------------------------------------------------------
extern "C" void kernel_launch(void* const* d_in, const int* in_sizes, int n_in,
                              void* d_out, int out_size, void* d_ws, size_t ws_size,
                              hipStream_t stream) {
  const float* x  = (const float*)d_in[0];   // [64,512,512]
  const float* h0 = (const float*)d_in[1];   // [64,1024]
  const float* c0 = (const float*)d_in[2];   // [64,1024]
  const float* Wx = (const float*)d_in[3];   // [512,4096]
  const float* Wh = (const float*)d_in[4];   // [1024,4096]
  const float* b  = (const float*)d_in[5];   // [4096]
  float* out = (float*)d_out;                // [64,1024]

  // Workspace layout (all rewritten from inputs every call -> deterministic):
  //   Wct  : 4096*1536 bf16 = 12 MB  (L2-resident across all 512 steps)
  //   xb   : 64*512*512 bf16 = 32 MB (x converted once)
  //   hbuf : 2 * 64*1024 bf16 (ping-pong)
  //   cbuf : 64*1024 fp32
  const size_t WCT_ELEMS  = (size_t)FOURH * KTOT;
  const size_t XB_ELEMS   = (size_t)B_ * T_ * I_;
  const size_t HBUF_ELEMS = (size_t)B_ * H_;
  char* ws = (char*)d_ws;
  unsigned short* Wct  = (unsigned short*)ws;                ws += WCT_ELEMS * 2;
  unsigned short* xb   = (unsigned short*)ws;                ws += XB_ELEMS * 2;
  unsigned short* hbuf = (unsigned short*)ws;                ws += 2 * HBUF_ELEMS * 2;
  float*          cbuf = (float*)ws;

  cvt_weights<<<(unsigned)((WCT_ELEMS + 255) / 256), 256, 0, stream>>>(Wx, Wh, Wct);
  cvt_x<<<(unsigned)((XB_ELEMS / 4 + 255) / 256), 256, 0, stream>>>(x, xb);
  init_state<<<(unsigned)((HBUF_ELEMS + 255) / 256), 256, 0, stream>>>(h0, c0, hbuf, cbuf);

  for (int t = 0; t < T_; ++t) {
    const unsigned short* hr = hbuf + (size_t)(t & 1) * HBUF_ELEMS;
    unsigned short*       hw = hbuf + (size_t)((t + 1) & 1) * HBUF_ELEMS;
    lstm_step<<<dim3(64, 4), 128, 0, stream>>>(xb, Wct, b, hr, hw, cbuf, out,
                                               t, t == T_ - 1 ? 1 : 0);
  }
}
